// NTM_13700945674927
// MI455X (gfx1250) — matrix operand order repaired and checked by reading, not compiled, not run on hardware
//
#include <hip/hip_runtime.h>
#include <math.h>

// NTM cell forward, MI455X (gfx1250, wave32).
// B=256, N=2048, M=64, H=512, R=2, WH=1, IN=128, OUT=128. EPS=1e-8.
//
// Pipeline (all on `stream`):
//  0a) pack_x_bf16 : Xb = bf16([inp | prev_reads^T | prev_h])   (256 x 768)
//  0b) pack_w_bf16 : Wb = bf16([W_ih | W_hh])                   (2048 x 768)
//  1) gates_wmma   : gates = Xb @ Wb^T + b via v_wmma_f32_16x16x32_bf16;
//                    inner loop is pure b128 loads + wmma (bf16 pre-packed)
//  2) lstm_pw      : c,h pointwise
//  3) head_params  : rp (2x256x70), wp (256x198) small GEMMs
//  4) addressing   : one block per (head,b): cosine sim + softmax + interp +
//                    circular shift + sharpen + renorm, all in LDS (1 mem pass)
//  5) mem_pass2    : fused reads (rbn,bnm->rbm) + erase/add write of new_mem
//                    (1 read + 1 write float4 sweep of the 134MB memory tensor)
//  6) out_gemm     : out = sigmoid([h|reads] @ out_W^T + out_b)

#define NTM_B   256
#define NTM_N   2048
#define NTM_M   64
#define NTM_H   512
#define NTM_R   2
#define NTM_IN  128
#define NTM_OUT 128
#define NTM_K1  768   // IN + R*M + H
#define NTM_4H  2048
#define NTM_EPS 1e-8f

typedef __attribute__((ext_vector_type(16))) __bf16        v16bf;
typedef __attribute__((ext_vector_type(8)))  float         v8f;
typedef __attribute__((ext_vector_type(4)))  unsigned int  u32x4;

union V16 { v16bf v; u32x4 q[2]; };

__device__ __forceinline__ unsigned short f2bf(float f) {
  unsigned int x = __float_as_uint(f);
  unsigned int r = x + 0x7FFFu + ((x >> 16) & 1u);   // round-to-nearest-even
  return (unsigned short)(r >> 16);
}
__device__ __forceinline__ float sigmoidf_(float x) { return 1.0f / (1.0f + expf(-x)); }
__device__ __forceinline__ float softplusf_(float x) { return fmaxf(x, 0.0f) + log1pf(expf(-fabsf(x))); }
__device__ __forceinline__ float dot4(float4 a, float4 b) {
  return a.x*b.x + a.y*b.y + a.z*b.z + a.w*b.w;
}

// ---------------- kernel 0a: pack LSTM input, f32 -> bf16 ----------------
__global__ void ntm_pack_x_bf16(const float* __restrict__ inp,
                                const float* __restrict__ prev_reads,
                                const float* __restrict__ prev_h,
                                unsigned short* __restrict__ Xb) {
  int i = blockIdx.x * blockDim.x + threadIdx.x;
  if (i >= NTM_B * NTM_K1) return;
  int b = i / NTM_K1, k = i % NTM_K1;
  float v;
  if (k < NTM_IN) {
    v = inp[b * NTM_IN + k];
  } else if (k < NTM_IN + NTM_R * NTM_M) {
    int t = k - NTM_IN; int r = t >> 6, m = t & 63;
    v = prev_reads[(r * NTM_B + b) * NTM_M + m];
  } else {
    v = prev_h[b * NTM_H + (k - NTM_IN - NTM_R * NTM_M)];
  }
  Xb[i] = f2bf(v);
}

// ---------------- kernel 0b: pack [W_ih | W_hh], f32 -> bf16 ----------------
__global__ void ntm_pack_w_bf16(const float* __restrict__ W_ih,
                                const float* __restrict__ W_hh,
                                unsigned short* __restrict__ Wb) {
  int i = blockIdx.x * blockDim.x + threadIdx.x;
  if (i >= NTM_4H * NTM_K1) return;
  int j = i / NTM_K1, k = i % NTM_K1;
  float v = (k < 256) ? W_ih[j * 256 + k] : W_hh[j * 512 + (k - 256)];
  Wb[i] = f2bf(v);
}

// ---------------- kernel 1: gates GEMM via WMMA bf16 ----------------
// gates(256x2048) = Xb(256x768) @ Wb^T + b. One wave -> 16(M) x 64(N) strip,
// K-loop 24 x 32. All fragment loads are contiguous 16B (b128).
__global__ __launch_bounds__(128) void ntm_gates_wmma(
    const unsigned short* __restrict__ Xb, const unsigned short* __restrict__ Wb,
    const float* __restrict__ bias, float* __restrict__ gates) {
  int wave = (blockIdx.x * blockDim.x + threadIdx.x) >> 5;  // 0..511
  int lane = threadIdx.x & 31;
  int b0 = (wave >> 5) * 16;        // 16 M-tiles
  int j0 = (wave & 31) * 64;        // 32 N-strips of 64
  int Mrow  = lane & 15;            // A row for this lane
  int khalf = (lane >> 4) * 16;     // A K-half select (elements)
  int Nc    = lane & 15;            // B/D column for this lane
  int kb    = (lane >> 4) * 16;     // B K-half select (elements)

  const unsigned short* xrow = Xb + (long)(b0 + Mrow) * NTM_K1 + khalf;
  const unsigned short* w0r  = Wb + (long)(j0 +  0 + Nc) * NTM_K1 + kb;
  const unsigned short* w1r  = Wb + (long)(j0 + 16 + Nc) * NTM_K1 + kb;
  const unsigned short* w2r  = Wb + (long)(j0 + 32 + Nc) * NTM_K1 + kb;
  const unsigned short* w3r  = Wb + (long)(j0 + 48 + Nc) * NTM_K1 + kb;

  v8f acc0 = {}, acc1 = {}, acc2 = {}, acc3 = {};

#pragma unroll 4
  for (int K0 = 0; K0 < NTM_K1; K0 += 32) {
    // A: element e -> K = khalf + e  => one contiguous 32B run
    V16 a;
    a.q[0] = *(const u32x4*)(xrow + K0);
    a.q[1] = *(const u32x4*)(xrow + K0 + 8);
    // B: element e -> K = kb + e  => one contiguous 32B run
    V16 bv0, bv1, bv2, bv3;
    bv0.q[0] = *(const u32x4*)(w0r + K0); bv0.q[1] = *(const u32x4*)(w0r + K0 + 8);
    bv1.q[0] = *(const u32x4*)(w1r + K0); bv1.q[1] = *(const u32x4*)(w1r + K0 + 8);
    bv2.q[0] = *(const u32x4*)(w2r + K0); bv2.q[1] = *(const u32x4*)(w2r + K0 + 8);
    bv3.q[0] = *(const u32x4*)(w3r + K0); bv3.q[1] = *(const u32x4*)(w3r + K0 + 8);

    acc0 = __builtin_amdgcn_wmma_f32_16x16x32_bf16(false, a.v, false, bv0.v, (short)0, acc0, false, false);
    acc1 = __builtin_amdgcn_wmma_f32_16x16x32_bf16(false, a.v, false, bv1.v, (short)0, acc1, false, false);
    acc2 = __builtin_amdgcn_wmma_f32_16x16x32_bf16(false, a.v, false, bv2.v, (short)0, acc2, false, false);
    acc3 = __builtin_amdgcn_wmma_f32_16x16x32_bf16(false, a.v, false, bv3.v, (short)0, acc3, false, false);
  }

  // D layout: VGPR v -> M = v + (lane>=16 ? 8 : 0), N = lane&15
  int Md = (lane >> 4) * 8;
#pragma unroll
  for (int v = 0; v < 8; ++v) {
    int row = b0 + Md + v;
    int c0 = j0 + Nc;
    gates[row * NTM_4H + c0     ] = acc0[v] + bias[c0     ];
    gates[row * NTM_4H + c0 + 16] = acc1[v] + bias[c0 + 16];
    gates[row * NTM_4H + c0 + 32] = acc2[v] + bias[c0 + 32];
    gates[row * NTM_4H + c0 + 48] = acc3[v] + bias[c0 + 48];
  }
}

// ---------------- kernel 2: LSTM pointwise ----------------
__global__ void ntm_lstm_pw(const float* __restrict__ gates,
                            const float* __restrict__ prev_c,
                            float* __restrict__ h_out, float* __restrict__ c_out) {
  int i = blockIdx.x * blockDim.x + threadIdx.x;
  if (i >= NTM_B * NTM_H) return;
  int b = i / NTM_H, j = i % NTM_H;
  const float* g = gates + b * NTM_4H;
  float ig = sigmoidf_(g[j]);
  float fg = sigmoidf_(g[NTM_H + j]);
  float gg = tanhf(g[2 * NTM_H + j]);
  float og = sigmoidf_(g[3 * NTM_H + j]);
  float c = fg * prev_c[i] + ig * gg;
  c_out[i] = c;
  h_out[i] = og * tanhf(c);
}

// ---------------- kernel 3: read/write head parameter GEMMs ----------------
__global__ void ntm_head_params(const float* __restrict__ h,
                                const float* __restrict__ read_W,
                                const float* __restrict__ read_b,
                                const float* __restrict__ write_W,
                                const float* __restrict__ write_b,
                                float* __restrict__ rp, float* __restrict__ wp) {
  const int PER_B = NTM_R * 70 + 198;  // 338
  int i = blockIdx.x * blockDim.x + threadIdx.x;
  if (i >= NTM_B * PER_B) return;
  int b = i / PER_B, o = i % PER_B;
  const float* wrow; float bias; float* dst;
  if (o < NTM_R * 70) {
    int r = o / 70, oo = o % 70;
    wrow = read_W + (r * 70 + oo) * NTM_H;
    bias = read_b[r * 70 + oo];
    dst  = rp + (r * NTM_B + b) * 70 + oo;
  } else {
    int oo = o - NTM_R * 70;
    wrow = write_W + oo * NTM_H;
    bias = write_b[oo];
    dst  = wp + b * 198 + oo;
  }
  const float4* h4 = (const float4*)(h + b * NTM_H);
  const float4* w4 = (const float4*)wrow;
  float acc = bias;
#pragma unroll 4
  for (int k = 0; k < NTM_H / 4; ++k) acc += dot4(h4[k], w4[k]);
  *dst = acc;
}

// ---------------- kernel 4: content+location addressing, 1 block = (head,b) ----------------
__global__ __launch_bounds__(256) void ntm_addressing(
    const float* __restrict__ rp, const float* __restrict__ wp,
    const float* __restrict__ prev_ws, const float* __restrict__ memory,
    float* __restrict__ weights_out) {
  int head = blockIdx.x >> 8;          // 0,1 = read; 2 = write
  int b    = blockIdx.x & 255;
  int tid  = threadIdx.x;

  __shared__ float k_s[NTM_M];
  __shared__ float par[7];             // kn, beta, g, s0, s1, s2, gamma
  __shared__ float l[NTM_N];           // logits / wc / wg staging (8 KB)
  __shared__ float red[256];

  const float* p = (head < NTM_R) ? (rp + (head * NTM_B + b) * 70) : (wp + b * 198);
  if (tid < NTM_M) k_s[tid] = tanhf(p[tid]);
  __syncthreads();
  if (tid == 0) {
    float s = 0.f;
    for (int m = 0; m < NTM_M; ++m) s += k_s[m] * k_s[m];
    par[0] = sqrtf(s);
    par[1] = softplusf_(p[NTM_M]);
    par[2] = sigmoidf_(p[NTM_M + 1]);
    float s0 = p[NTM_M + 2], s1 = p[NTM_M + 3], s2 = p[NTM_M + 4];
    float mx = fmaxf(s0, fmaxf(s1, s2));
    float e0 = expf(s0 - mx), e1 = expf(s1 - mx), e2 = expf(s2 - mx);
    float es = e0 + e1 + e2;
    par[3] = e0 / es; par[4] = e1 / es; par[5] = e2 / es;
    par[6] = 1.0f + softplusf_(p[NTM_M + 5]);
  }
  __syncthreads();
  float kn = par[0], beta = par[1];

  // cosine-sim logits (streams mem[b] once, float4)
  const float* memb = memory + (long)b * NTM_N * NTM_M;
  const float4* kv4 = (const float4*)k_s;
  for (int it = 0; it < NTM_N / 256; ++it) {
    int n = it * 256 + tid;
    const float4* m4 = (const float4*)(memb + (long)n * NTM_M);
    float dot = 0.f, nrm = 0.f;
#pragma unroll
    for (int q = 0; q < NTM_M / 4; ++q) {
      float4 v = m4[q], kq = kv4[q];
      dot += dot4(v, kq);
      nrm += dot4(v, v);
    }
    l[n] = beta * (dot / (kn * sqrtf(nrm) + NTM_EPS));
  }
  __syncthreads();

  // softmax over n
  float mymax = -3.4e38f;
  for (int it = 0; it < NTM_N / 256; ++it) mymax = fmaxf(mymax, l[it * 256 + tid]);
  red[tid] = mymax; __syncthreads();
  for (int s = 128; s > 0; s >>= 1) { if (tid < s) red[tid] = fmaxf(red[tid], red[tid + s]); __syncthreads(); }
  float mx = red[0]; __syncthreads();
  float mysum = 0.f;
  for (int it = 0; it < NTM_N / 256; ++it) {
    int n = it * 256 + tid;
    float e = expf(l[n] - mx);
    l[n] = e; mysum += e;
  }
  red[tid] = mysum; __syncthreads();
  for (int s = 128; s > 0; s >>= 1) { if (tid < s) red[tid] += red[tid + s]; __syncthreads(); }
  float sum = red[0]; __syncthreads();

  // interpolate with previous weights
  float g = par[2], s0 = par[3], s1 = par[4], s2 = par[5], gamma = par[6];
  const float* pwv = prev_ws + ((long)head * NTM_B + b) * NTM_N;
  for (int it = 0; it < NTM_N / 256; ++it) {
    int n = it * 256 + tid;
    float wc = l[n] / sum;
    l[n] = g * wc + (1.0f - g) * pwv[n];
  }
  __syncthreads();

  // circular shift + sharpen + renormalize
  float wt_loc[NTM_N / 256];
  float mys2 = 0.f;
  for (int it = 0; it < NTM_N / 256; ++it) {
    int n = it * 256 + tid;
    float wt = s0 * l[(n + NTM_N - 1) & (NTM_N - 1)] + s1 * l[n] + s2 * l[(n + 1) & (NTM_N - 1)];
    float wpow = expf(gamma * logf(wt + NTM_EPS));
    wt_loc[it] = wpow; mys2 += wpow;
  }
  red[tid] = mys2; __syncthreads();
  for (int s = 128; s > 0; s >>= 1) { if (tid < s) red[tid] += red[tid + s]; __syncthreads(); }
  float inv = 1.0f / (red[0] + NTM_EPS);
  float* wout = weights_out + ((long)head * NTM_B + b) * NTM_N;
  for (int it = 0; it < NTM_N / 256; ++it) wout[it * 256 + tid] = wt_loc[it] * inv;
}

// ---------------- kernel 5: fused reads + erase/add memory write ----------------
// 1 block = 1 batch row; 256 threads = 16 n-rows x 16 float4 m-lanes. Every
// memory op is b128; one read + one write sweep of the 134MB memory tensor.
__global__ __launch_bounds__(256) void ntm_mem_pass2(
    const float* __restrict__ memory, const float* __restrict__ weights,
    const float* __restrict__ wp, float* __restrict__ new_mem,
    float* __restrict__ reads_out) {
  int b = blockIdx.x;
  int tid = threadIdx.x;
  int lm = tid & 15;      // float4 lane within a row (m = 4*lm .. 4*lm+3)
  int ng = tid >> 4;      // 16 n-rows per iteration

  __shared__ float e_s[NTM_M], a_s[NTM_M];
  __shared__ float4 red4[2][256];
  if (tid < NTM_M) {
    e_s[tid] = sigmoidf_(wp[b * 198 + 70 + tid]);
    a_s[tid] = wp[b * 198 + 134 + tid];
  }
  __syncthreads();

  const float* memb = memory + (long)b * NTM_N * NTM_M;
  float* nmb        = new_mem + (long)b * NTM_N * NTM_M;
  const float* w0 = weights + ((long)0 * NTM_B + b) * NTM_N;
  const float* w1 = weights + ((long)1 * NTM_B + b) * NTM_N;
  const float* ww = weights + ((long)2 * NTM_B + b) * NTM_N;

  float4 e4 = ((const float4*)e_s)[lm];
  float4 a4 = ((const float4*)a_s)[lm];
  float4 acc0 = {0.f, 0.f, 0.f, 0.f};
  float4 acc1 = {0.f, 0.f, 0.f, 0.f};

  for (int n0 = 0; n0 < NTM_N; n0 += 16) {
    int n = n0 + ng;
    float4 v = *(const float4*)(memb + (long)n * NTM_M + 4 * lm);
    float w0n = w0[n], w1n = w1[n], wn = ww[n];
    acc0.x += w0n * v.x; acc0.y += w0n * v.y; acc0.z += w0n * v.z; acc0.w += w0n * v.w;
    acc1.x += w1n * v.x; acc1.y += w1n * v.y; acc1.z += w1n * v.z; acc1.w += w1n * v.w;
    float4 nm;
    nm.x = v.x * (1.0f - wn * e4.x) + wn * a4.x;
    nm.y = v.y * (1.0f - wn * e4.y) + wn * a4.y;
    nm.z = v.z * (1.0f - wn * e4.z) + wn * a4.z;
    nm.w = v.w * (1.0f - wn * e4.w) + wn * a4.w;
    *(float4*)(nmb + (long)n * NTM_M + 4 * lm) = nm;
  }
  red4[0][tid] = acc0; red4[1][tid] = acc1; __syncthreads();
  if (tid < 16) {
    float4 r0 = {0.f, 0.f, 0.f, 0.f};
    float4 r1 = {0.f, 0.f, 0.f, 0.f};
    for (int gset = 0; gset < 16; ++gset) {
      float4 t0 = red4[0][gset * 16 + tid];
      float4 t1 = red4[1][gset * 16 + tid];
      r0.x += t0.x; r0.y += t0.y; r0.z += t0.z; r0.w += t0.w;
      r1.x += t1.x; r1.y += t1.y; r1.z += t1.z; r1.w += t1.w;
    }
    *(float4*)(reads_out + (0 * NTM_B + b) * NTM_M + 4 * tid) = r0;
    *(float4*)(reads_out + (1 * NTM_B + b) * NTM_M + 4 * tid) = r1;
  }
}

// ---------------- kernel 6: output projection ----------------
__global__ void ntm_out_gemm(const float* __restrict__ h,
                             const float* __restrict__ reads,
                             const float* __restrict__ out_W,
                             const float* __restrict__ out_b,
                             float* __restrict__ out) {
  int i = blockIdx.x * blockDim.x + threadIdx.x;
  if (i >= NTM_B * NTM_OUT) return;
  int b = i / NTM_OUT, o = i % NTM_OUT;
  const float* wrow = out_W + (long)o * (NTM_H + NTM_R * NTM_M);
  float acc = out_b[o];
  const float4* h4 = (const float4*)(h + b * NTM_H);
  const float4* w4 = (const float4*)wrow;
#pragma unroll 4
  for (int k = 0; k < NTM_H / 4; ++k) acc += dot4(h4[k], w4[k]);
#pragma unroll
  for (int r = 0; r < NTM_R; ++r) {
    const float4* r4  = (const float4*)(reads + (r * NTM_B + b) * NTM_M);
    const float4* w4b = (const float4*)(wrow + NTM_H + r * NTM_M);
#pragma unroll
    for (int k = 0; k < NTM_M / 4; ++k) acc += dot4(r4[k], w4b[k]);
  }
  out[i] = sigmoidf_(acc);
}

extern "C" void kernel_launch(void* const* d_in, const int* in_sizes, int n_in,
                              void* d_out, int out_size, void* d_ws, size_t ws_size,
                              hipStream_t stream) {
  const float* inp        = (const float*)d_in[0];
  const float* prev_reads = (const float*)d_in[1];
  const float* prev_h     = (const float*)d_in[2];
  const float* prev_c     = (const float*)d_in[3];
  const float* prev_ws    = (const float*)d_in[4];
  const float* memory     = (const float*)d_in[5];
  const float* W_ih       = (const float*)d_in[6];
  const float* W_hh       = (const float*)d_in[7];
  const float* b_lstm     = (const float*)d_in[8];
  const float* read_W     = (const float*)d_in[9];
  const float* read_b     = (const float*)d_in[10];
  const float* write_W    = (const float*)d_in[11];
  const float* write_b    = (const float*)d_in[12];
  const float* out_W      = (const float*)d_in[13];
  const float* out_b      = (const float*)d_in[14];

  // d_out: out | reads | h | c | weights | new_mem (flat, float32)
  float* out     = (float*)d_out;
  float* o_out   = out;                // 256*128      = 32768
  float* o_reads = out + 32768;        // 2*256*64     = 32768
  float* o_h     = out + 65536;        // 256*512      = 131072
  float* o_c     = out + 196608;       // 256*512      = 131072
  float* o_w     = out + 327680;       // 3*256*2048   = 1572864
  float* o_mem   = out + 1900544;      // 256*2048*64  = 33554432

  // workspace (float slots):
  //   Xb   (bf16 256x768)   =  98304
  //   Wb   (bf16 2048x768)  = 786432
  //   gates(f32  256x2048)  = 524288
  //   rp   (f32  2x256x70)  =  35840
  //   wp   (f32  256x198)   =  50688     total ~6.0 MB
  float*          ws    = (float*)d_ws;
  unsigned short* ws_xb = (unsigned short*)ws;              // 98304 float slots
  unsigned short* ws_wb = (unsigned short*)(ws + 98304);    // 786432 float slots
  float*          ws_g  = ws + 884736;
  float*          ws_rp = ws + 1409024;
  float*          ws_wp = ws + 1444864;

  ntm_pack_x_bf16<<<768, 256, 0, stream>>>(inp, prev_reads, prev_h, ws_xb);
  ntm_pack_w_bf16<<<6144, 256, 0, stream>>>(W_ih, W_hh, ws_wb);
  ntm_gates_wmma <<<128, 128, 0, stream>>>(ws_xb, ws_wb, b_lstm, ws_g);
  ntm_lstm_pw    <<<512, 256, 0, stream>>>(ws_g, prev_c, o_h, o_c);
  ntm_head_params<<<338, 256, 0, stream>>>(o_h, read_W, read_b, write_W, write_b, ws_rp, ws_wp);
  ntm_addressing <<<768, 256, 0, stream>>>(ws_rp, ws_wp, prev_ws, memory, o_w);
  ntm_mem_pass2  <<<256, 256, 0, stream>>>(memory, o_w, ws_wp, o_mem, o_reads);
  ntm_out_gemm   <<<128, 256, 0, stream>>>(o_h, o_reads, out_W, out_b, o_out);
}
